// WindowCrossBandDir_8564164789081
// MI455X (gfx1250) — matrix operand order repaired and checked by reading, not compiled, not run on hardware
//
#include <hip/hip_runtime.h>
#include <hip/hip_bf16.h>

typedef __attribute__((ext_vector_type(16))) _Float16 v16h;
typedef __attribute__((ext_vector_type(8)))  _Float16 v8h;
typedef __attribute__((ext_vector_type(8)))  float    v8f;

#define LDH 72   // padded leading dim (halves): 144B row stride, 16B aligned, 36-bank stride

union V16U { v16h v; struct { v8h lo, hi; } s; };

// ---- WMMA operand loaders: all contiguous per-lane (2x b128) ----

// A tile (16x32) from row-major [m][k] f16 (LDS or global). Lane reads halves
// {k0+kh+0..7} and {k0+kh+16..23} of row m  (kh = 8*(lane>>4)).
__device__ inline v16h load_a_rm(const _Float16* __restrict__ src, int ld, int row0, int k0) {
  const int l = threadIdx.x & 31;
  const _Float16* p = src + (row0 + (l & 15)) * ld + k0 + ((l >> 4) << 3);
  V16U u;
  u.s.lo = *(const v8h*)p;
  u.s.hi = *(const v8h*)(p + 16);
  return u.v;
}

// A tile with only k<16 valid (head slice): load halves {kb+kh..kb+kh+7}, upper k zero.
__device__ inline v16h load_a_klo(const _Float16* __restrict__ src, int ld, int row0, int kb) {
  const int l = threadIdx.x & 31;
  const _Float16* p = src + (row0 + (l & 15)) * ld + kb + ((l >> 4) << 3);
  V16U u;
  v8h z = {};
  u.s.lo = *(const v8h*)p;
  u.s.hi = z;
  return u.v;
}

// B tile (32x16) from column-major storage [col][k] (i.e. transposed), element (k, col).
// Lane reads halves {k0+kh+0..7} and {k0+kh+16..23} of row col (kh = 8*(lane>>4)).
__device__ inline v16h load_b_ct(const _Float16* __restrict__ src, int ld, int k0, int col0) {
  const int l = threadIdx.x & 31;
  const _Float16* p = src + (col0 + (l & 15)) * ld + k0 + ((l >> 4) << 3);
  V16U u;
  u.s.lo = *(const v8h*)p;
  u.s.hi = *(const v8h*)(p + 16);
  return u.v;
}

// B tile with only k<16 valid: load halves {kb+kh..kb+kh+7}, upper k zero.
__device__ inline v16h load_b_klo(const _Float16* __restrict__ src, int ld, int kb, int col0) {
  const int l = threadIdx.x & 31;
  V16U u;
  {
    const _Float16* p = src + (col0 + (l & 15)) * ld + kb + ((l >> 4) << 3);
    v8h z = {};
    u.s.lo = *(const v8h*)p;
    u.s.hi = z;
  }
  return u.v;
}

// Store C (16x16 f32) transposed into [col][row] f16: per-lane 8 contiguous halves (b128).
__device__ inline void store_c_t(_Float16* __restrict__ dst, int ld, int row0, int col0, v8f c) {
  const int l = threadIdx.x & 31;
  v8h h;
#pragma unroll
  for (int r = 0; r < 8; ++r) h[r] = (_Float16)c[r];
  _Float16* p = dst + (col0 + (l & 15)) * ld + row0 + ((l >> 4) << 3);
  *(v8h*)p = h;
}

// Store C row-major [row][col] f16 (scatter b16, conflict-free: lanes contiguous in col).
__device__ inline void store_c_rm(_Float16* __restrict__ dst, int ld, int row0, int col0, v8f c) {
  const int l = threadIdx.x & 31;
  const int n = col0 + (l & 15);
  const int mo = (l >> 4) * 8;
#pragma unroll
  for (int r = 0; r < 8; ++r) dst[(row0 + mo + r) * ld + n] = (_Float16)c[r];
}

// 64x64x64 GEMM: P = W(f16 global, [m][k] ld 64) @ X(LDS, [n][k] ld LDH).
// TOUT: true -> store P transposed [n][m]; false -> row-major [m][n].
template <bool TOUT>
__device__ inline void proj_gemm(const _Float16* __restrict__ W, const _Float16* __restrict__ X,
                                 _Float16* __restrict__ P, int wv) {
#pragma unroll
  for (int t = 2 * wv; t < 2 * wv + 2; ++t) {
    const int mt = (t >> 2) * 16, nt = (t & 3) * 16;
    v8f acc = {};
#pragma unroll
    for (int kc = 0; kc < 64; kc += 32) {
      v16h a = load_a_rm(W, 64, mt, kc);
      v16h b = load_b_ct(X, LDH, kc, nt);
      acc = __builtin_amdgcn_wmma_f32_16x16x32_f16(false, a, false, b, (short)0, acc, false, false);
    }
    if (TOUT) store_c_t(P, LDH, mt, nt, acc);
    else      store_c_rm(P, LDH, mt, nt, acc);
  }
}

// #windows covering row r (ws=8, ss=4, starts 0..120).
__device__ inline float covcnt(int r) {
  int lo = r - 7; if (lo < 0) lo = 0; lo = (lo + 3) & ~3;
  int hi = (r < 120) ? r : 120; hi &= ~3;
  return (float)((hi - lo) / 4 + 1);
}

// ---------------- Kernel A1: per-channel global mean of `high` ----------------
__global__ void __launch_bounds__(256) se_pool_kernel(const float* __restrict__ high,
                                                      float* __restrict__ pooled) {
  __shared__ float red[256];
  const int bc = blockIdx.x;
  const float* src = high + ((size_t)bc << 14);
  float s = 0.f;
  for (int i = threadIdx.x; i < 16384; i += 256) s += src[i];
  red[threadIdx.x] = s;
  __syncthreads();
  for (int st = 128; st > 0; st >>= 1) {
    if (threadIdx.x < st) red[threadIdx.x] += red[threadIdx.x + st];
    __syncthreads();
  }
  if (threadIdx.x == 0) pooled[bc] = red[0] * (1.0f / 16384.0f);
}

// ---------------- Kernel A2: squeeze-excite gates ----------------
__global__ void se_gate_kernel(const float* __restrict__ pooled,
                               const float* __restrict__ w10, const float* __restrict__ w20,
                               const float* __restrict__ w11, const float* __restrict__ w21,
                               const float* __restrict__ w12, const float* __restrict__ w22,
                               float* __restrict__ s_se) {
  const int t = threadIdx.x;
  if (t >= 12) return;
  const int b = t / 3, g = t % 3;
  const int gc  = (g == 0) ? 22 : 21;
  const int ofs = (g == 0) ? 0 : ((g == 1) ? 22 : 43);
  const float* w1 = (g == 0) ? w10 : ((g == 1) ? w11 : w12);
  const float* w2 = (g == 0) ? w20 : ((g == 1) ? w21 : w22);
  float acc = 0.f;
  for (int j = 0; j < gc; ++j) acc += pooled[b * 64 + ofs + j] * w1[j];
  acc = fmaxf(acc, 0.f);
  for (int j = 0; j < gc; ++j) {
    const float x = acc * w2[j];
    s_se[b * 64 + ofs + j] = 1.0f / (1.0f + __expf(-x));
  }
}

// ---------------- Kernel A3: convert all 8 weight matrices to f16 once ----------------
__global__ void __launch_bounds__(256) wconv_kernel(
    const float* __restrict__ w0, const float* __restrict__ w1, const float* __restrict__ w2,
    const float* __restrict__ w3, const float* __restrict__ w4, const float* __restrict__ w5,
    const float* __restrict__ w6, const float* __restrict__ w7, _Float16* __restrict__ dst) {
  const int w = blockIdx.x;
  const float* src = (w == 0) ? w0 : (w == 1) ? w1 : (w == 2) ? w2 : (w == 3) ? w3
                   : (w == 4) ? w4 : (w == 5) ? w5 : (w == 6) ? w6 : w7;
  _Float16* d = dst + w * 4096;
  for (int i = threadIdx.x; i < 4096; i += 256) d[i] = (_Float16)src[i];
}

// ---------------- Kernel B: per-window cross attention (both directions) ----------------
__global__ void __launch_bounds__(256) window_attn_kernel(
    const float* __restrict__ low, const float* __restrict__ high,
    const float* __restrict__ s_se, const _Float16* __restrict__ wf16,
    float* __restrict__ accL, float* __restrict__ accH) {
  __shared__ _Float16 sXl[64 * LDH];  // low window,  [pos][chan]
  __shared__ _Float16 sXh[64 * LDH];  // high window, [pos][chan] (SE-gated)
  __shared__ _Float16 sQt[64 * LDH];  // Q projection transposed: [pos][chan]
  __shared__ _Float16 sKt[64 * LDH];  // K projection transposed: [pos][chan]
  __shared__ _Float16 sV [64 * LDH];  // V projection: [chan][pos]
  __shared__ _Float16 sS [64 * LDH];  // logits -> probs, [n][m]

  const int tid = threadIdx.x;
  const int wv  = tid >> 5;
  const int blk = blockIdx.x;
  const int b   = blk / (31 * 31);
  const int r0  = blk % (31 * 31);
  const int wy  = r0 / 31, wx = r0 % 31;

  // Coalesced global read (pos fastest), transposed LDS write (2-way conflict w/ LDH=72).
  for (int i = tid; i < 4096; i += 256) {
    const int c = i >> 6, n = i & 63;
    const int row = wy * 4 + (n >> 3), col = wx * 4 + (n & 7);
    const size_t g = ((size_t)(b * 64 + c) * 128 + row) * 128 + col;
    sXl[n * LDH + c] = (_Float16)low[g];
    sXh[n * LDH + c] = (_Float16)(high[g] * s_se[b * 64 + c]);
  }
  __syncthreads();

  for (int dir = 0; dir < 2; ++dir) {
    const _Float16* wq  = wf16 + (dir ? 3 : 0) * 4096;
    const _Float16* wk  = wf16 + (dir ? 4 : 1) * 4096;
    const _Float16* wvv = wf16 + (dir ? 5 : 2) * 4096;
    const _Float16* Xq  = dir ? sXh : sXl;
    const _Float16* Xkv = dir ? sXl : sXh;
    float* acc_img = dir ? accH : accL;

    proj_gemm<true >(wq,  Xq,  sQt, wv);   // Q^T: [pos][chan]
    proj_gemm<true >(wk,  Xkv, sKt, wv);   // K^T: [pos][chan]
    proj_gemm<false>(wvv, Xkv, sV,  wv);   // V:   [chan][pos]
    __syncthreads();

    for (int h = 0; h < 4; ++h) {
      const int dbase = h * 16;
      // logits S[n,m] = sum_{d<16} Q[d,n] K[d,m]  (K-dim padded to 32 with zeros)
#pragma unroll
      for (int t = 2 * wv; t < 2 * wv + 2; ++t) {
        const int nt = (t >> 2) * 16, mt = (t & 3) * 16;
        v16h a  = load_a_klo(sQt, LDH, nt, dbase);
        v16h bm = load_b_klo(sKt, LDH, dbase, mt);
        v8f s = {};
        s = __builtin_amdgcn_wmma_f32_16x16x32_f16(false, a, false, bm, (short)0, s, false, false);
        store_c_rm(sS, LDH, nt, mt, s);
      }
      __syncthreads();
      // row softmax (scale = 1/sqrt(16) = 0.25), in place on own row
      if (tid < 64) {
        float mx = -1e30f;
        for (int m2 = 0; m2 < 64; ++m2)
          mx = fmaxf(mx, 0.25f * (float)sS[tid * LDH + m2]);
        float sum = 0.f;
        for (int m2 = 0; m2 < 64; ++m2) {
          const float e = __expf(0.25f * (float)sS[tid * LDH + m2] - mx);
          sum += e;
          sS[tid * LDH + m2] = (_Float16)e;
        }
        const float inv = 1.0f / sum;
        for (int m2 = 0; m2 < 64; ++m2)
          sS[tid * LDH + m2] = (_Float16)((float)sS[tid * LDH + m2] * inv);
      }
      __syncthreads();
      // O^T[n, dd] = sum_m A[n,m] * V[dbase+dd, m]; fold via atomics
      if (wv < 4) {
        const int nt = wv * 16;
        v8f o = {};
#pragma unroll
        for (int kc = 0; kc < 64; kc += 32) {
          v16h a  = load_a_rm(sS, LDH, nt, kc);
          v16h bm = load_b_ct(sV, LDH, kc, dbase);
          o = __builtin_amdgcn_wmma_f32_16x16x32_f16(false, a, false, bm, (short)0, o, false, false);
        }
        const int l  = tid & 31;
        const int dd = l & 15, mo = (l >> 4) * 8;
#pragma unroll
        for (int r = 0; r < 8; ++r) {
          const int n = nt + mo + r;
          const int row = wy * 4 + (n >> 3), col = wx * 4 + (n & 7);
          atomicAdd(&acc_img[((size_t)(b * 64 + dbase + dd) * 128 + row) * 128 + col], o[r]);
        }
      }
      __syncthreads();
    }
    __syncthreads();
  }
}

// ---------------- Kernel C: divide by overlap count, final 1x1 proj, residual ----------------
__global__ void __launch_bounds__(256) final_proj_kernel(
    const float* __restrict__ accL, const float* __restrict__ accH,
    const float* __restrict__ low, const float* __restrict__ high,
    const _Float16* __restrict__ wf16, float* __restrict__ out) {
  __shared__ _Float16 sB[128 * LDH];  // [pixel][chan]
  const int tid = threadIdx.x, wv = tid >> 5;
  const int pt  = blockIdx.x;
  const int bb  = blockIdx.y;
  const int dir = blockIdx.z;
  const float* acc = dir ? accH : accL;
  const float* res = dir ? high : low;
  const _Float16* W = wf16 + (dir ? 7 : 6) * 4096;
  float* dst = out + (size_t)dir * 4194304;
  const int pix0 = pt * 128;

  for (int i = tid; i < 64 * 128; i += 256) {
    const int c = i >> 7, p = i & 127;
    const int pix = pix0 + p;
    const float inv = 1.0f / (covcnt(pix >> 7) * covcnt(pix & 127));
    sB[p * LDH + c] = (_Float16)(acc[((size_t)(bb * 64 + c) << 14) + pix] * inv);
  }
  __syncthreads();

  // 4 m-tiles x 8 n-tiles = 32 tiles; 4 per wave
  for (int t = 4 * wv; t < 4 * wv + 4; ++t) {
    const int mt = (t >> 3) * 16, nt = (t & 7) * 16;
    v8f o = {};
#pragma unroll
    for (int kc = 0; kc < 64; kc += 32) {
      v16h a  = load_a_rm(W, 64, mt, kc);
      v16h bm = load_b_ct(sB, LDH, kc, nt);
      o = __builtin_amdgcn_wmma_f32_16x16x32_f16(false, a, false, bm, (short)0, o, false, false);
    }
    const int l = tid & 31;
    const int n = nt + (l & 15), mo = (l >> 4) * 8;
#pragma unroll
    for (int r = 0; r < 8; ++r) {
      const int oc = mt + mo + r;
      const size_t g = ((size_t)(bb * 64 + oc) << 14) + pix0 + n;
      dst[g] = res[g] + o[r];
    }
  }
}

extern "C" void kernel_launch(void* const* d_in, const int* in_sizes, int n_in,
                              void* d_out, int out_size, void* d_ws, size_t ws_size,
                              hipStream_t stream) {
  const float* low      = (const float*)d_in[0];
  const float* high     = (const float*)d_in[1];
  const float* w_ql     = (const float*)d_in[2];
  const float* w_kh     = (const float*)d_in[3];
  const float* w_vh     = (const float*)d_in[4];
  const float* w_qh     = (const float*)d_in[5];
  const float* w_kl     = (const float*)d_in[6];
  const float* w_vl     = (const float*)d_in[7];
  const float* w_proj_l = (const float*)d_in[8];
  const float* w_proj_h = (const float*)d_in[9];

  // workspace layout
  float*    pooled = (float*)d_ws;                                  // 256 f
  float*    s_se   = pooled + 256;                                  // 256 f
  _Float16* wf16   = (_Float16*)((char*)d_ws + 2048);               // 8*4096 f16 = 64 KB
  float*    accL   = (float*)((char*)d_ws + 2048 + 65536);          // 4*64*128*128 f
  float*    accH   = accL + 4194304;

  hipMemsetAsync(accL, 0, 2ull * 4194304 * sizeof(float), stream);

  se_pool_kernel<<<256, 256, 0, stream>>>(high, pooled);
  se_gate_kernel<<<1, 32, 0, stream>>>(pooled,
      (const float*)d_in[10], (const float*)d_in[11],
      (const float*)d_in[12], (const float*)d_in[13],
      (const float*)d_in[14], (const float*)d_in[15], s_se);
  wconv_kernel<<<8, 256, 0, stream>>>(w_ql, w_kh, w_vh, w_qh, w_kl, w_vl,
                                      w_proj_l, w_proj_h, wf16);

  window_attn_kernel<<<4 * 31 * 31, 256, 0, stream>>>(low, high, s_se, wf16, accL, accH);

  dim3 gC(128, 4, 2);
  final_proj_kernel<<<gC, 256, 0, stream>>>(accL, accH, low, high, wf16, (float*)d_out);
}
